// MotionPredictor_35579509080358
// MI455X (gfx1250) — matrix operand order repaired and checked by reading, not compiled, not run on hardware
//
#include <hip/hip_runtime.h>
#include <hip/hip_bf16.h>

// ---------------------------------------------------------------------------
// MotionPredictor for MI455X (gfx1250, wave32, WMMA 16x16x32 f16 / f32-acc)
// LSTM recurrent GEMMs stream weights through LDS with async global->LDS
// copies (double-buffered K-slices) to avoid the register-spill path.
// ---------------------------------------------------------------------------

typedef __attribute__((ext_vector_type(16))) _Float16 v16h;
typedef __attribute__((ext_vector_type(8)))  float    v8f;

#define B_    8
#define T_    32
#define HID_  256
#define G_    1024          // 4*HID
#define PIX_  (512*512)
#define SLICE_HALFS (64*32*16)   // one kt-slice: 64 ntiles * 32 lanes * 16 halves
#define SLICE_BYTES (SLICE_HALFS*2)

__device__ __forceinline__ v8f wmma_f16(v16h a, v16h b, v8f c) {
  return __builtin_amdgcn_wmma_f32_16x16x32_f16(false, a, false, b, (short)0, c,
                                                false, false);
}

__device__ __forceinline__ float sigm(float x) { return 1.f / (1.f + __expf(-x)); }

// --- gfx1250 async global->LDS copy (16B per lane), tracked by ASYNCcnt ----
__device__ __forceinline__ void async_ld16(unsigned lds_off, const void* gaddr) {
  asm volatile("global_load_async_to_lds_b128 %0, %1, off"
               :: "v"(lds_off), "v"(gaddr) : "memory");
}
__device__ __forceinline__ void wait_async0() {
  asm volatile("s_wait_asynccnt 0x0" ::: "memory");
}
__device__ __forceinline__ unsigned lds_off_of(const void* p) {
  return (unsigned)(size_t)p;   // low 32 bits of LDS aperture addr = LDS offset
}
// issue one 64KB kt-slice: 256 threads * 16 rounds * 16B = 65536 B
__device__ __forceinline__ void issue_slice(const _Float16* __restrict__ Bw,
                                            int kt, unsigned bufBase, int tid) {
  const char* src = (const char*)Bw + (size_t)kt * SLICE_BYTES + (size_t)tid * 16;
  const unsigned dst = bufBase + (unsigned)tid * 16u;
#pragma unroll
  for (int r = 0; r < 16; ++r)
    async_ld16(dst + (unsigned)r * 4096u, src + (size_t)r * 4096);
}

// ---------------------------------------------------------------------------
// Kernel 1: per-(b,t) centroid of 512x512 prediction map.  Bandwidth bound:
// 268 MB at 23.3 TB/s.  One block per image, float4 (b128) coalesced loads.
// ---------------------------------------------------------------------------
__global__ __launch_bounds__(1024)
void k_centroid(const float* __restrict__ pred, float* __restrict__ loc) {
  const int bt = blockIdx.x;                      // 0..255 = b*T + t
  const float4* img4 = (const float4*)(pred + (size_t)bt * PIX_);
  float s = 0.f, sx = 0.f, sy = 0.f;
  for (int i = threadIdx.x; i < PIX_ / 4; i += 1024) {
    float4 v = img4[i];
    float row  = (float)(i >> 7);
    float colb = (float)((i & 127) << 2);
    float rs = (v.x + v.y) + (v.z + v.w);
    s  += rs;
    sy += rs * row;
    sx += v.x * colb + v.y * (colb + 1.f) + v.z * (colb + 2.f) + v.w * (colb + 3.f);
  }
  __shared__ float r0[1024], r1[1024], r2[1024];
  r0[threadIdx.x] = s; r1[threadIdx.x] = sx; r2[threadIdx.x] = sy;
  __syncthreads();
  for (int off = 512; off > 0; off >>= 1) {
    if (threadIdx.x < (unsigned)off) {
      r0[threadIdx.x] += r0[threadIdx.x + off];
      r1[threadIdx.x] += r1[threadIdx.x + off];
      r2[threadIdx.x] += r2[threadIdx.x + off];
    }
    __syncthreads();
  }
  if (threadIdx.x == 0) {
    float ps = fmaxf(r0[0], 1e-8f);
    loc[bt * 2 + 0] = r1[0] / ps;                 // cx
    loc[bt * 2 + 1] = r2[0] / ps;                 // cy
  }
}

// ---------------------------------------------------------------------------
// Kernel 2: prep.
//  - xg0[b*T+t][n] = cx*Wih0[n][0] + cy*Wih0[n][1] + bih0[n] + bhh0[n]
//  - convert Whh0 / Whh1 / Wih1 ([1024][256] row-major) to f16 in slice-major
//    WMMA B layout:  Bbuf[((kt*64+nt)*32+lane)*16+hi] =
//       W[nt*16 + (lane&15)][kt*32 + (lane>=16?16:0) + hi]
// ---------------------------------------------------------------------------
__global__ __launch_bounds__(256)
void k_prep(const float* __restrict__ Wih0, const float* __restrict__ bih0,
            const float* __restrict__ bhh0,
            const float* __restrict__ Whh0, const float* __restrict__ Whh1,
            const float* __restrict__ Wih1,
            const float* __restrict__ loc,
            float* __restrict__ xg0,
            _Float16* __restrict__ Whh0h, _Float16* __restrict__ Whh1h,
            _Float16* __restrict__ Wih1h) {
  const int idx = blockIdx.x * 256 + threadIdx.x;     // 0 .. 262143
  {
    const int n  = idx & (G_ - 1);
    const int bt = idx >> 10;
    xg0[idx] = loc[bt * 2 + 0] * Wih0[n * 2 + 0]
             + loc[bt * 2 + 1] * Wih0[n * 2 + 1]
             + bih0[n] + bhh0[n];
  }
  {
    const int hi   = idx & 15;
    const int lane = (idx >> 4) & 31;
    const int nt   = (idx >> 9) & 63;
    const int kt   = idx >> 15;
    const int n    = nt * 16 + (lane & 15);
    const int k    = kt * 32 + ((lane >= 16) ? 16 : 0) + hi;
    const int src  = n * HID_ + k;
    Whh0h[idx] = (_Float16)Whh0[src];
    Whh1h[idx] = (_Float16)Whh1[src];
    Wih1h[idx] = (_Float16)Wih1[src];
  }
}

// A-layout position for element (row m, k) of a 16x32 f16 A tile
__device__ __forceinline__ void a_pos(int m, int k, int& kt, int& lane, int& hi) {
  kt = k >> 5;
  const int kin = k & 31;
  const int grp = kin >> 4;
  const int within = kin & 15;
  lane = (grp << 4) + m;
  hi   = within;
}

// ---------------------------------------------------------------------------
// Kernel 3: LSTM layer 0 recurrence. 1 block, 8 waves. Weights streamed
// through LDS (double-buffered 64KB kt-slices, async copies). Wave w owns
// N-tiles w*8..w*8+7. Emits out1 in f16 A-layout for the layer-1 input GEMM.
// ---------------------------------------------------------------------------
__global__ __launch_bounds__(256)
void k_lstm0(const float* __restrict__ xg0, const _Float16* __restrict__ Bw,
             _Float16* __restrict__ out1A) {
  __shared__ _Float16 buf[2][SLICE_HALFS];  // 128 KB weight slices
  __shared__ float    g[8 * G_];            // 32 KB gates (rows = batch)
  __shared__ _Float16 hA[8 * 32 * 16];      // 8 KB h in WMMA A layout
  const int tid  = threadIdx.x;
  const int lane = tid & 31;
  const int wave = tid >> 5;
  const int m    = tid >> 5;
  const int k0   = (tid & 31) * 8;
  const unsigned bufOff0 = lds_off_of(&buf[0][0]);
  const unsigned bufOff1 = lds_off_of(&buf[1][0]);

  for (int i = tid; i < 8 * 32 * 16; i += 256) hA[i] = (_Float16)0.f;
  float c[8];
#pragma unroll
  for (int j = 0; j < 8; ++j) c[j] = 0.f;
  issue_slice(Bw, 0, bufOff0, tid);         // prefetch slice 0 for t=0

  for (int t = 0; t < T_; ++t) {
    v8f acc[8];
#pragma unroll
    for (int nt = 0; nt < 8; ++nt) acc[nt] = (v8f){};
    for (int kt = 0; kt < 8; ++kt) {
      wait_async0();
      __syncthreads();                      // slice kt landed for all waves
      if (kt < 7)
        issue_slice(Bw, kt + 1, (kt & 1) ? bufOff0 : bufOff1, tid);
      v16h a = *(const v16h*)&hA[(kt * 32 + lane) * 16];
      const _Float16* bp = &buf[kt & 1][0];
#pragma unroll
      for (int nt = 0; nt < 8; ++nt) {
        const int ntg = wave * 8 + nt;
        v16h b = *(const v16h*)&bp[(ntg * 32 + lane) * 16];
        acc[nt] = wmma_f16(a, b, acc[nt]);
      }
      __syncthreads();                      // all waves done reading buf[kt&1]
    }
    if (lane < 16) {                        // batch rows 0..7 in lanes 0..15
#pragma unroll
      for (int nt = 0; nt < 8; ++nt) {
        const int ntg = wave * 8 + nt;
#pragma unroll
        for (int r = 0; r < 8; ++r) g[r * G_ + ntg * 16 + lane] = acc[nt][r];
      }
    }
    __syncthreads();

    const float* xgr = xg0 + (size_t)(m * T_ + t) * G_;
#pragma unroll
    for (int j = 0; j < 8; ++j) {
      const int k = k0 + j;
      float gi = g[m * G_ + k]            + xgr[k];
      float gf = g[m * G_ + HID_ + k]     + xgr[HID_ + k];
      float gg = g[m * G_ + 2 * HID_ + k] + xgr[2 * HID_ + k];
      float go = g[m * G_ + 3 * HID_ + k] + xgr[3 * HID_ + k];
      const float i_ = sigm(gi), f_ = sigm(gf), g_ = tanhf(gg), o_ = sigm(go);
      c[j] = f_ * c[j] + i_ * g_;
      const float h_ = o_ * tanhf(c[j]);
      int kt2, l2, hh2;
      a_pos(m, k, kt2, l2, hh2);
      hA[(kt2 * 32 + l2) * 16 + hh2] = (_Float16)h_;
      const int row = m * T_ + t;           // out1 in A-layout, row = b*T+t
      const int mt = row >> 4, mr = row & 15;
      int kt3, l3, hh3;
      a_pos(mr, k, kt3, l3, hh3);
      out1A[((size_t)(mt * 8 + kt3) * 32 + l3) * 16 + hh3] = (_Float16)h_;
    }
    if (t + 1 < T_) issue_slice(Bw, 0, bufOff0, tid);   // prefetch next step
    __syncthreads();                        // hA ready for next step
  }
}

// ---------------------------------------------------------------------------
// Kernel 4: xg1 = out1 @ Wih1^T + (bih1 + bhh1).  M=256, N=1024, K=256.
// One wave per 16x16 output tile: 1024 waves = 128 blocks.
// ---------------------------------------------------------------------------
__global__ __launch_bounds__(256)
void k_gemm_xg1(const _Float16* __restrict__ A, const _Float16* __restrict__ Bw,
                const float* __restrict__ bih1, const float* __restrict__ bhh1,
                float* __restrict__ xg1) {
  const int wid  = blockIdx.x * 8 + (threadIdx.x >> 5);
  const int lane = threadIdx.x & 31;
  const int mt   = wid >> 6;              // 0..15
  const int nt   = wid & 63;              // 0..63
  v8f acc = (v8f){};
#pragma unroll
  for (int kt = 0; kt < 8; ++kt) {
    v16h a = *(const v16h*)&A[((size_t)(mt * 8 + kt) * 32 + lane) * 16];
    v16h b = *(const v16h*)&Bw[((size_t)(kt * 64 + nt) * 32 + lane) * 16];
    acc = wmma_f16(a, b, acc);
  }
  const int col  = nt * 16 + (lane & 15);
  const float bias = bih1[col] + bhh1[col];
  const int mbase = mt * 16 + ((lane < 16) ? 0 : 8);
#pragma unroll
  for (int r = 0; r < 8; ++r)
    xg1[(size_t)(mbase + r) * G_ + col] = acc[r] + bias;
}

// ---------------------------------------------------------------------------
// Kernel 5: LSTM layer 1 recurrence + MLP head + cumsum.  1 block, 8 waves.
// ---------------------------------------------------------------------------
__global__ __launch_bounds__(256)
void k_lstm1_head(const float* __restrict__ xg1, const _Float16* __restrict__ Bw,
                  const float* __restrict__ loc,
                  const float* __restrict__ W1, const float* __restrict__ b1,
                  const float* __restrict__ W2, const float* __restrict__ b2,
                  float* __restrict__ out) {
  __shared__ _Float16 buf[2][SLICE_HALFS];  // 128 KB
  __shared__ float    g[8 * G_];            // 32 KB
  __shared__ _Float16 hA[8 * 32 * 16];      // 8 KB
  __shared__ float    hf[B_ * HID_];        // final hidden
  __shared__ float    hid[B_ * HID_];       // relu layer
  __shared__ float    pm[B_ * 10];
  const int tid  = threadIdx.x;
  const int lane = tid & 31;
  const int wave = tid >> 5;
  const int m    = tid >> 5;
  const int k0   = (tid & 31) * 8;
  const unsigned bufOff0 = lds_off_of(&buf[0][0]);
  const unsigned bufOff1 = lds_off_of(&buf[1][0]);
  float* mf = out + 160;                    // motion_features region

  for (int i = tid; i < 8 * 32 * 16; i += 256) hA[i] = (_Float16)0.f;
  float c[8];
#pragma unroll
  for (int j = 0; j < 8; ++j) c[j] = 0.f;
  issue_slice(Bw, 0, bufOff0, tid);

  for (int t = 0; t < T_; ++t) {
    v8f acc[8];
#pragma unroll
    for (int nt = 0; nt < 8; ++nt) acc[nt] = (v8f){};
    for (int kt = 0; kt < 8; ++kt) {
      wait_async0();
      __syncthreads();
      if (kt < 7)
        issue_slice(Bw, kt + 1, (kt & 1) ? bufOff0 : bufOff1, tid);
      v16h a = *(const v16h*)&hA[(kt * 32 + lane) * 16];
      const _Float16* bp = &buf[kt & 1][0];
#pragma unroll
      for (int nt = 0; nt < 8; ++nt) {
        const int ntg = wave * 8 + nt;
        v16h b = *(const v16h*)&bp[(ntg * 32 + lane) * 16];
        acc[nt] = wmma_f16(a, b, acc[nt]);
      }
      __syncthreads();
    }
    if (lane < 16) {
#pragma unroll
      for (int nt = 0; nt < 8; ++nt) {
        const int ntg = wave * 8 + nt;
#pragma unroll
        for (int r = 0; r < 8; ++r) g[r * G_ + ntg * 16 + lane] = acc[nt][r];
      }
    }
    __syncthreads();

    const float* xgr = xg1 + (size_t)(m * T_ + t) * G_;
#pragma unroll
    for (int j = 0; j < 8; ++j) {
      const int k = k0 + j;
      float gi = g[m * G_ + k]            + xgr[k];
      float gf = g[m * G_ + HID_ + k]     + xgr[HID_ + k];
      float gg = g[m * G_ + 2 * HID_ + k] + xgr[2 * HID_ + k];
      float go = g[m * G_ + 3 * HID_ + k] + xgr[3 * HID_ + k];
      const float i_ = sigm(gi), f_ = sigm(gf), g_ = tanhf(gg), o_ = sigm(go);
      c[j] = f_ * c[j] + i_ * g_;
      const float h_ = o_ * tanhf(c[j]);
      int kt2, l2, hh2;
      a_pos(m, k, kt2, l2, hh2);
      hA[(kt2 * 32 + l2) * 16 + hh2] = (_Float16)h_;
      hf[m * HID_ + k] = h_;
      mf[(size_t)(m * T_ + t) * HID_ + k] = h_;   // motion_features (B,T,H)
    }
    if (t + 1 < T_) issue_slice(Bw, 0, bufOff0, tid);
    __syncthreads();
  }

  // ---- head: hid = relu(h_last @ W1^T + b1) ----
#pragma unroll
  for (int j = 0; j < 8; ++j) {
    const int n = k0 + j;
    float s = b1[n];
    for (int k = 0; k < HID_; ++k) s += hf[m * HID_ + k] * W1[n * HID_ + k];
    hid[m * HID_ + n] = fmaxf(s, 0.f);
  }
  __syncthreads();
  if (tid < 80) {                           // pm = hid @ W2^T + b2 (8 x 10)
    const int b = tid / 10, idx = tid % 10;
    float s = b2[idx];
    for (int k = 0; k < HID_; ++k) s += hid[b * HID_ + k] * W2[idx * HID_ + k];
    pm[tid] = s;
  }
  __syncthreads();
  if (tid < 16) {                           // motion + cumsum + last_location
    const int b = tid >> 1, cc = tid & 1;
    float run = loc[(b * T_ + (T_ - 1)) * 2 + cc];
#pragma unroll
    for (int p = 0; p < 5; ++p) {
      const float mo = pm[b * 10 + p * 2 + cc];
      out[b * 10 + p * 2 + cc] = mo;
      run += mo;
      out[80 + b * 10 + p * 2 + cc] = run;
    }
  }
}

// ---------------------------------------------------------------------------
extern "C" void kernel_launch(void* const* d_in, const int* in_sizes, int n_in,
                              void* d_out, int out_size, void* d_ws, size_t ws_size,
                              hipStream_t stream) {
  const float* predictions = (const float*)d_in[1];
  const float* Wih0 = (const float*)d_in[3];
  const float* Whh0 = (const float*)d_in[4];
  const float* bih0 = (const float*)d_in[5];
  const float* bhh0 = (const float*)d_in[6];
  const float* Wih1 = (const float*)d_in[7];
  const float* Whh1 = (const float*)d_in[8];
  const float* bih1 = (const float*)d_in[9];
  const float* bhh1 = (const float*)d_in[10];
  const float* W1   = (const float*)d_in[11];
  const float* b1   = (const float*)d_in[12];
  const float* W2   = (const float*)d_in[13];
  const float* b2   = (const float*)d_in[14];
  float* out = (float*)d_out;

  char* ws = (char*)d_ws;
  float*    loc   = (float*)(ws + 0);               //   2 KB
  float*    xg0   = (float*)(ws + 2048);            //   1 MB
  float*    xg1   = (float*)(ws + 2048 + 1048576);  //   1 MB
  _Float16* Whh0h = (_Float16*)(ws + 2099200);      // 512 KB (slice-major)
  _Float16* Whh1h = (_Float16*)(ws + 2623488);      // 512 KB
  _Float16* Wih1h = (_Float16*)(ws + 3147776);      // 512 KB
  _Float16* out1A = (_Float16*)(ws + 3672064);      // 128 KB

  k_centroid<<<B_ * T_, 1024, 0, stream>>>(predictions, loc);
  k_prep<<<(B_ * T_ * G_) / 256, 256, 0, stream>>>(Wih0, bih0, bhh0, Whh0, Whh1,
                                                   Wih1, loc, xg0, Whh0h, Whh1h,
                                                   Wih1h);
  k_lstm0<<<1, 256, 0, stream>>>(xg0, Whh0h, out1A);
  k_gemm_xg1<<<128, 256, 0, stream>>>(out1A, Wih1h, bih1, bhh1, xg1);
  k_lstm1_head<<<1, 256, 0, stream>>>(xg1, Whh1h, loc, W1, b1, W2, b2, out);
}
